// MoEBlock_77163382440429
// MI455X (gfx1250) — compile-verified, matrix operand order checked by reading, not run on hardware
//
#include <hip/hip_runtime.h>
#include <hip/hip_bf16.h>
#include <float.h>

// Problem constants (from setup_inputs): N=8192 tokens, D=1024, H=1024, E=16, k=2.
#define TOK_N 8192
#define DIM_D 1024
#define DIM_H 1024
#define NUM_E 16
#define KT 32              // WMMA K depth per step (bf16 16x16x32)
#define HT 128             // H columns per block (8 waves x 16)
#define NKSTEP (DIM_D / KT)

typedef __attribute__((ext_vector_type(16))) __bf16 v16bf;
typedef __attribute__((ext_vector_type(8)))  float  v8f;

union V16 { uint4 q[2]; v16bf v; };

static __device__ __forceinline__ unsigned short bf16_hi_rn(float f) {
  unsigned int u = __float_as_uint(f);
  u += 0x7FFFu + ((u >> 16) & 1u);       // round-to-nearest-even
  return (unsigned short)(u >> 16);
}
static __device__ __forceinline__ float bf16_to_f32(unsigned short h) {
  return __uint_as_float(((unsigned int)h) << 16);
}
static __device__ __forceinline__ unsigned pack_hi(float f0, float f1) {
  return ((unsigned)bf16_hi_rn(f1) << 16) | bf16_hi_rn(f0);
}
static __device__ __forceinline__ unsigned pack_lo(float f0, float f1) {
  unsigned short h0 = bf16_hi_rn(f0), h1 = bf16_hi_rn(f1);
  unsigned short l0 = bf16_hi_rn(f0 - bf16_to_f32(h0));
  unsigned short l1 = bf16_hi_rn(f1 - bf16_to_f32(h1));
  return ((unsigned)l1 << 16) | l0;
}

// ---------------------------------------------------------------------------
// Precompute 1: x -> plain row-major bf16 hi/lo.  One dword (2 elems) / thread.
// ---------------------------------------------------------------------------
__global__ __launch_bounds__(256)
void convert_x_bf16(const float* __restrict__ x,
                    unsigned* __restrict__ xhi, unsigned* __restrict__ xlo) {
  const size_t q = (size_t)blockIdx.x * 256 + threadIdx.x;   // [0, N*D/2)
  const float2 f = *(const float2*)(x + 2 * q);
  xhi[q] = pack_hi(f.x, f.y);
  xlo[q] = pack_lo(f.x, f.y);
}

// ---------------------------------------------------------------------------
// Precompute 2: expert_w -> bf16 hi/lo, PRE-SWIZZLED into the CDNA5 16-bit
// B-fragment layout: for (e, kblk, hgroup): 32 lanes x 8 dwords contiguous.
//   lane = (h%16) + 16*((K%16)>=8), dword v = 4*(K>=16) + (K%8)/2, half = K%2.
// Thread q = ((((e*32+kb)*64)+hg)*32+lane)*8 + v  -> fully coalesced writes.
// ---------------------------------------------------------------------------
__global__ __launch_bounds__(256)
void convert_w_swizzle(const float* __restrict__ ew,
                       unsigned* __restrict__ whi, unsigned* __restrict__ wlo) {
  const unsigned q = blockIdx.x * 256 + threadIdx.x;         // [0, E*D*H/2)
  const int v    = q & 7;
  const int lane = (q >> 3) & 31;
  const int hg   = (q >> 8) & 63;
  const int kb   = (q >> 14) & 31;
  const int e    = q >> 19;
  const int K0 = ((lane >> 4) << 3) + ((v & 4) << 2) + ((v & 3) << 1);
  const int d  = (kb << 5) + K0;                             // K0 even; pair d,d+1
  const int h  = (hg << 4) + (lane & 15);
  const size_t src = ((size_t)e * DIM_D + d) * DIM_H + h;
  const float f0 = ew[src];
  const float f1 = ew[src + DIM_H];
  whi[q] = pack_hi(f0, f1);
  wlo[q] = pack_lo(f0, f1);
}

// ---------------------------------------------------------------------------
// Kernel 1: router logits -> softmax -> top-2 -> per-expert compacted lists.
// One wave32 per token; wave32-safe shuffles only.
// ---------------------------------------------------------------------------
__global__ __launch_bounds__(256)
void moe_router_topk(const float* __restrict__ x,
                     const float* __restrict__ rw,
                     const float* __restrict__ rb,
                     int* __restrict__ counts,
                     int* __restrict__ tok_idx,
                     float* __restrict__ tok_w) {
  const int tok  = (blockIdx.x * blockDim.x + threadIdx.x) >> 5;
  const int lane = threadIdx.x & 31;
  if (tok >= TOK_N) return;

  const int e    = lane & 15;
  const int half = lane >> 4;

  const float* xr = x + (size_t)tok * DIM_D + (size_t)half * (DIM_D / 2);
  const float* wr = rw + (size_t)half * (DIM_D / 2) * NUM_E + e;
  float acc = 0.0f;
#pragma unroll 4
  for (int d = 0; d < DIM_D / 2; ++d)
    acc = fmaf(xr[d], wr[(size_t)d * NUM_E], acc);
  acc += __shfl_xor(acc, 16, 32);
  acc += rb[e];

  float m = acc;
#pragma unroll
  for (int s = 8; s >= 1; s >>= 1) m = fmaxf(m, __shfl_xor(m, s, 32));
  float p = __expf(acc - m);
  float ssum = p;
#pragma unroll
  for (int s = 8; s >= 1; s >>= 1) ssum += __shfl_xor(ssum, s, 32);
  const float w = p / ssum;

  float v1 = w; int i1 = e;
#pragma unroll
  for (int s = 8; s >= 1; s >>= 1) {
    float ov = __shfl_xor(v1, s, 32);
    int   oi = __shfl_xor(i1, s, 32);
    if (ov > v1 || (ov == v1 && oi < i1)) { v1 = ov; i1 = oi; }
  }
  float v2 = (e == i1) ? -FLT_MAX : w; int i2 = e;
#pragma unroll
  for (int s = 8; s >= 1; s >>= 1) {
    float ov = __shfl_xor(v2, s, 32);
    int   oi = __shfl_xor(i2, s, 32);
    if (ov > v2 || (ov == v2 && oi < i2)) { v2 = ov; i2 = oi; }
  }

  if (lane == 0) {
    int p1 = atomicAdd(&counts[i1], 1);
    tok_idx[i1 * TOK_N + p1] = tok;
    tok_w  [i1 * TOK_N + p1] = v1;
    int p2 = atomicAdd(&counts[i2], 1);
    tok_idx[i2 * TOK_N + p2] = tok;
    tok_w  [i2 * TOK_N + p2] = v2;
  }
}

// ---------------------------------------------------------------------------
// Kernel 2: per-expert GEMM, bf16 hi/lo split WMMA, double-buffered async-to-LDS
// staging (ASYNCcnt). Each wave issues exactly 5 async ops per K-tile:
// 1x A (b64) + 4x B (b128); s_wait_asynccnt 5 => previous tile complete.
// ---------------------------------------------------------------------------
__global__ __launch_bounds__(256)
void moe_expert_gemm(const unsigned short* __restrict__ xhi,
                     const unsigned short* __restrict__ xlo,
                     const unsigned short* __restrict__ whi,
                     const unsigned short* __restrict__ wlo,
                     const float* __restrict__ eb,
                     const int*   __restrict__ counts,
                     const int*   __restrict__ tok_idx,
                     const float* __restrict__ tok_w,
                     float* __restrict__ out) {
  const int e    = blockIdx.z;
  const int cnt  = counts[e];
  const int row0 = blockIdx.y * 16;
  if (row0 >= cnt) return;                        // block-uniform early exit
  const int valid = (cnt - row0 < 16) ? (cnt - row0) : 16;
  const int hbase = blockIdx.x * HT;

  const int tid  = threadIdx.x;
  const int lane = tid & 31;
  const int wv   = tid >> 5;

  // A: 32 "rows" (16 hi + 16 lo) x 64B of K data, 80B padded stride
  // B: 8KB hi | 8KB lo, already in fragment order (lane*32B contiguous)
  __shared__ __align__(16) unsigned char sA[2][32 * 80];
  __shared__ __align__(16) unsigned char sB[2][16384];
  __shared__ int   sIdx[16];
  __shared__ float sW[16];

  if (tid < 16) {
    const int rr = row0 + ((tid < valid) ? tid : 0);
    sIdx[tid] = tok_idx[e * TOK_N + rr];
    sW[tid]   = (tid < valid) ? tok_w[e * TOK_N + rr] : 0.0f;
  }
  __syncthreads();

  // Per-thread staging tasks (uniform: every thread issues 1 A + 4 B asyncs).
  const int rr    = tid >> 3;                      // 0..31 (row: 0-15 hi, 16-31 lo)
  const int c8    = tid & 7;                       // 8B chunk within 64B row
  const unsigned short* aSrc0 =
      ((rr >> 4) ? xlo : xhi) + (size_t)sIdx[rr & 15] * DIM_D + c8 * 4;
  const unsigned aLdsOff = (unsigned)(rr * 80 + c8 * 8);

  auto stage = [&](int kb, int buf) {
    {
      const unsigned long long g = (unsigned long long)(aSrc0 + (size_t)kb * KT);
      const unsigned l = (unsigned)(size_t)&sA[buf][0] + aLdsOff;
      asm volatile("global_load_async_to_lds_b64 %0, %1, off"
                   :: "v"(l), "v"(g) : "memory");
    }
    const size_t baseDW = (((size_t)e * 32 + kb) * 64 + (hbase >> 4)) * 256;
#pragma unroll
    for (int j = 0; j < 4; ++j) {                  // j<2: hi, j>=2: lo (uniform)
      const int q    = tid + 256 * j;              // 0..1023
      const int isLo = q >> 9;
      const int ql   = q & 511;                    // 16B task within hi/lo half
      const unsigned short* src =
          (isLo ? wlo : whi) + baseDW * 2 + (size_t)ql * 8;
      const unsigned long long g = (unsigned long long)src;
      const unsigned l =
          (unsigned)(size_t)&sB[buf][0] + (unsigned)(isLo * 8192 + ql * 16);
      asm volatile("global_load_async_to_lds_b128 %0, %1, off"
                   :: "v"(l), "v"(g) : "memory");
    }
  };

  v8f c = {0.f, 0.f, 0.f, 0.f, 0.f, 0.f, 0.f, 0.f};
  const int arow = (lane & 15) * 80 + (lane >> 4) * 16;

  stage(0, 0);
  for (int i = 0; i < NKSTEP; ++i) {
    const int p = i & 1;
    if (i + 1 < NKSTEP) {
      stage(i + 1, p ^ 1);                         // prefetch next tile
      asm volatile("s_wait_asynccnt 0x5" ::: "memory");  // tile i landed
    } else {
      asm volatile("s_wait_asynccnt 0x0" ::: "memory");
    }
    __syncthreads();                               // publish LDS across waves

    const unsigned char* bufA = &sA[p][0];
    const unsigned char* bufB = &sB[p][0];
    V16 ahi, alo, bhi, blo;
    ahi.q[0] = *(const uint4*)(bufA + arow);
    ahi.q[1] = *(const uint4*)(bufA + arow + 32);
    alo.q[0] = *(const uint4*)(bufA + 16 * 80 + arow);
    alo.q[1] = *(const uint4*)(bufA + 16 * 80 + arow + 32);
    const unsigned char* bp = bufB + wv * 1024 + lane * 32;
    bhi.q[0] = *(const uint4*)(bp);
    bhi.q[1] = *(const uint4*)(bp + 16);
    blo.q[0] = *(const uint4*)(bp + 8192);
    blo.q[1] = *(const uint4*)(bp + 8192 + 16);

    // D += Ahi*Bhi + Ahi*Blo + Alo*Bhi   (~fp32-accurate product)
    c = __builtin_amdgcn_wmma_f32_16x16x32_bf16(false, ahi.v, false, bhi.v,
                                                (short)0, c, false, false);
    c = __builtin_amdgcn_wmma_f32_16x16x32_bf16(false, ahi.v, false, blo.v,
                                                (short)0, c, false, false);
    c = __builtin_amdgcn_wmma_f32_16x16x32_bf16(false, alo.v, false, bhi.v,
                                                (short)0, c, false, false);
    __syncthreads();                               // readers done before overwrite
  }

  // out[n,h] += w * (acc + bias); exactly 2 commutative fp32 adds per element.
  const int Nc   = lane & 15;
  const int mofs = (lane >> 4) * 8;                // C layout: M = v + 8*(lane>=16)
  const int h    = hbase + wv * 16 + Nc;
#pragma unroll
  for (int v = 0; v < 8; ++v) {
    const int M = v + mofs;
    if (M < valid) {
      const float wt  = sW[M];
      const float val = wt * (c[v] + eb[e * DIM_H + h]);
      atomicAdd(&out[(size_t)sIdx[M] * DIM_H + h], val);
    }
  }
}

// ---------------------------------------------------------------------------
extern "C" void kernel_launch(void* const* d_in, const int* in_sizes, int n_in,
                              void* d_out, int out_size, void* d_ws, size_t ws_size,
                              hipStream_t stream) {
  const float* x  = (const float*)d_in[0];   // [N, D]
  const float* rw = (const float*)d_in[1];   // [D, E]
  const float* rb = (const float*)d_in[2];   // [E]
  const float* ew = (const float*)d_in[3];   // [E, D, H]
  const float* eb = (const float*)d_in[4];   // [E, H]
  // d_in[5] is k (== 2), baked into the top-2 router.

  unsigned char* w = (unsigned char*)d_ws;
  int*   counts  = (int*)w;                                   // 64 B
  int*   tok_idx = (int*)(w + 256);                           // 512 KB
  float* tok_w   = (float*)(w + 256 + (size_t)NUM_E * TOK_N * 4);
  unsigned short* xhi = (unsigned short*)(w + (1u << 20));    // 16 MB
  unsigned short* xlo = xhi + (size_t)TOK_N * DIM_D;          // 16 MB
  unsigned short* whi = xlo + (size_t)TOK_N * DIM_D;          // 32 MB
  unsigned short* wlo = whi + (size_t)NUM_E * DIM_D * DIM_H;  // 32 MB
  float* out = (float*)d_out;

  hipMemsetAsync(counts, 0, NUM_E * sizeof(int), stream);
  hipMemsetAsync(out, 0, (size_t)out_size * sizeof(float), stream);

  convert_x_bf16<<<TOK_N * DIM_D / 2 / 256, 256, 0, stream>>>(
      x, (unsigned*)xhi, (unsigned*)xlo);
  convert_w_swizzle<<<NUM_E * DIM_D * DIM_H / 2 / 256, 256, 0, stream>>>(
      ew, (unsigned*)whi, (unsigned*)wlo);

  moe_router_topk<<<TOK_N / 8, 256, 0, stream>>>(x, rw, rb, counts, tok_idx, tok_w);

  dim3 grid(DIM_H / HT, TOK_N / 16, NUM_E);
  moe_expert_gemm<<<grid, 256, 0, stream>>>(xhi, xlo, whi, wlo, eb,
                                            counts, tok_idx, tok_w, out);
}
